// MongeAmpereFlow_14259291422969
// MI455X (gfx1250) — matrix operand order repaired and checked by reading, not compiled, not run on hardware
//
#include <hip/hip_runtime.h>
#include <hip/hip_bf16.h>

typedef __attribute__((ext_vector_type(16))) _Float16 v16h;
typedef __attribute__((ext_vector_type(8)))  _Float16 v8h;
typedef __attribute__((ext_vector_type(8)))  float    v8f;

union Frag { v16h v; v8h h[2]; };

#define MAF_EPS   0.1f
#define MAF_STEPS 2
#define MAF_SIGN  1.0f
#define NLOG2E    (-1.4426950408889634f)   // -log2(e)

// One wave owns 16 samples for the full 2-step RK4 flow.
// State layout (per lane): value[nt*8+r] holds sample m = r + 8*(lane/16),
// dim/col = nt*16 + (lane%16) -- identical to the WMMA f32 C/D tile layout,
// so GEMM outputs drop straight into the RK4 state registers.
//
// Algebraic folds:
//   GEMM1 B = w1 * (-log2 e)  -> WMMA output is directly the exp2 argument
//   GEMM2 B = w1 * w2         -> stage raw sigmoid s, no per-element w2 mul
//   bias    : sigmoid(z+b) = 1/(1 + e^{-b} * 2^{-z log2e}), e^{-b} in LDS
__global__ __launch_bounds__(128) void maflow_rk4_wmma(
    const float* __restrict__ x, const float* __restrict__ logp,
    const float* __restrict__ w1, const float* __restrict__ b1,
    const float* __restrict__ w2, const float* __restrict__ b2,
    float* __restrict__ xout, float* __restrict__ lpout)
{
  constexpr int D = 32, H = 256;
  __shared__ __align__(16) _Float16 sW1c[H * D];   // col-major [h][d]: w1 * (-log2e)   (GEMM1 B)
  __shared__ __align__(16) _Float16 sW1w2[D * H];  // row-major [d][h]: w1 * w2         (GEMM2 B)
  __shared__ float sEB[H];                         // exp(-b1[h])
  __shared__ float sW2C6[H];                       // (1/6) * w2[h] * ||w1[:,h]||^2
  __shared__ float sW2C3[H];                       // (1/3) * w2[h] * ||w1[:,h]||^2
  __shared__ __align__(16) _Float16 sStage[4][16 * 32];  // per-wave A-transpose staging
  __shared__ __align__(16) float    sRed[4][16][16];     // per-wave logp reduction

  const int tid = threadIdx.x;

  // ---- one-time weight staging (f32 -> f16, scaled, both layouts) ----
  for (int i = tid; i < D * H; i += 128) {
    const int d = i >> 8, h = i & (H - 1);
    const float v = w1[i];
    sW1w2[i]        = (_Float16)(v * w2[h]);
    sW1c[h * D + d] = (_Float16)(v * NLOG2E);
  }
  for (int h = tid; h < H; h += 128) {
    float c = 0.f;
    for (int d = 0; d < D; ++d) { const float v = w1[d * H + h]; c += v * v; }
    const float wc = w2[h] * c;        // Laplacian column factor
    sW2C6[h] = wc * (1.f / 6.f);
    sW2C3[h] = wc * (1.f / 3.f);
    sEB[h]   = __expf(-b1[h]);
  }
  __syncthreads();

  const int wave = tid >> 5;
  const int lane = tid & 31;
  const int half = lane >> 4;
  const int lm   = lane & 15;
  const int row0 = (blockIdx.x * 4 + wave) * 16;
  _Float16* stage = sStage[wave];

  // ---- load x tile into RK4 state registers ----
  float xc[16];
#pragma unroll
  for (int nt = 0; nt < 2; ++nt)
#pragma unroll
    for (int r = 0; r < 8; ++r)
      xc[nt * 8 + r] = x[(size_t)(row0 + r + 8 * half) * D + nt * 16 + lm];

  float lpacc[8];
#pragma unroll
  for (int r = 0; r < 8; ++r) lpacc[r] = 0.f;

  const float h1 = MAF_SIGN * MAF_EPS;
  const float h2 = 0.5f * h1;
  const float stagec[4] = {0.f, h2, h2, h1};
  const float wgt[4]    = {1.f / 6.f, 1.f / 3.f, 1.f / 3.f, 1.f / 6.f};

  float g[16];
  for (int step = 0; step < MAF_STEPS; ++step) {
    float xacc[16];
#pragma unroll
    for (int i = 0; i < 16; ++i) { xacc[i] = 0.f; g[i] = 0.f; }

#pragma unroll
    for (int e = 0; e < 4; ++e) {
      const float pc = stagec[e];
      const float we = wgt[e];
      // RK4 Laplacian weight class: 1/6 for stages 0,3 ; 1/3 for stages 1,2
      const float* __restrict__ sWeC = (e == 0 || e == 3) ? sW2C6 : sW2C3;

      // stage point p = xc + pc*g  ->  f16 row-major [m][k] staging (transpose via LDS)
#pragma unroll
      for (int nt = 0; nt < 2; ++nt)
#pragma unroll
        for (int r = 0; r < 8; ++r) {
          const float pv = __builtin_fmaf(pc, g[nt * 8 + r], xc[nt * 8 + r]);
          stage[(r + 8 * half) * 32 + nt * 16 + lm] = (_Float16)pv;
        }
      asm volatile("s_wait_dscnt 0x0" ::: "memory");  // wave-local cross-lane RAW fence

      Frag ax;  // A fragment of P (16x32 f16): row = lm, contiguous K
      ax.h[0] = *(const v8h*)&stage[lm * 32 + half * 16];
      ax.h[1] = *(const v8h*)&stage[lm * 32 + half * 16 + 8];
      asm volatile("" ::: "memory");

      v8f g0, g1, czero;
#pragma unroll
      for (int r = 0; r < 8; ++r) { g0[r] = 0.f; g1[r] = 0.f; czero[r] = 0.f; }

      for (int j = 0; j < 8; ++j) {           // 32 hidden cols per iteration
#pragma unroll
        for (int t = 0; t < 2; ++t) {
          const int hcol = (2 * j + t) * 16 + lm;
          Frag bw;  // B fragment of (-log2e * w1)[32 x 16tile], contiguous K
          bw.h[0] = *(const v8h*)&sW1c[hcol * D + half * 16];
          bw.h[1] = *(const v8h*)&sW1c[hcol * D + half * 16 + 8];
          // zz = -log2e * (P @ w1_tile) : exp2 argument straight out of the WMMA
          v8f zz = __builtin_amdgcn_wmma_f32_16x16x32_f16(
              false, ax.v, false, bw.v, (short)0, czero, false, false);
          const float eb  = sEB[hcol];
          const float wec = sWeC[hcol];
#pragma unroll
          for (int r = 0; r < 8; ++r) {
            // sigmoid(z+b) = 1 / (1 + e^{-b} 2^{zz}) : v_exp + v_fma + v_rcp
            const float ez = __builtin_amdgcn_exp2f(zz[r]);
            const float s  = __builtin_amdgcn_rcpf(__builtin_fmaf(eb, ez, 1.f));
            const float q  = __builtin_fmaf(-s, s, s);          // s(1-s)
            lpacc[r] = __builtin_fmaf(q, wec, lpacc[r]);        // weighted Laplacian
            stage[(r + 8 * half) * 32 + t * 16 + lm] = (_Float16)s;
          }
        }
        asm volatile("s_wait_dscnt 0x0" ::: "memory");
        Frag ag;  // A fragment of S (16x32 chunk j)
        ag.h[0] = *(const v8h*)&stage[lm * 32 + half * 16];
        ag.h[1] = *(const v8h*)&stage[lm * 32 + half * 16 + 8];
        asm volatile("" ::: "memory");
        // B fragments of (w1*w2)^T chunk: B2[k][d] = w1[d][32j+k]*w2[32j+k]
        Frag bd0, bd1;
        bd0.h[0] = *(const v8h*)&sW1w2[(size_t)lm * H + 32 * j + half * 16];
        bd0.h[1] = *(const v8h*)&sW1w2[(size_t)lm * H + 32 * j + half * 16 + 8];
        bd1.h[0] = *(const v8h*)&sW1w2[(size_t)(16 + lm) * H + 32 * j + half * 16];
        bd1.h[1] = *(const v8h*)&sW1w2[(size_t)(16 + lm) * H + 32 * j + half * 16 + 8];
        g0 = __builtin_amdgcn_wmma_f32_16x16x32_f16(false, ag.v, false, bd0.v, (short)0, g0, false, false);
        g1 = __builtin_amdgcn_wmma_f32_16x16x32_f16(false, ag.v, false, bd1.v, (short)0, g1, false, false);
      }

#pragma unroll
      for (int r = 0; r < 8; ++r) { g[r] = g0[r]; g[8 + r] = g1[r]; }
#pragma unroll
      for (int i = 0; i < 16; ++i) xacc[i] = __builtin_fmaf(we, g[i], xacc[i]);
    }
#pragma unroll
    for (int i = 0; i < 16; ++i) xc[i] = __builtin_fmaf(h1, xacc[i], xc[i]);
  }

  // ---- write x out ----
#pragma unroll
  for (int nt = 0; nt < 2; ++nt)
#pragma unroll
    for (int r = 0; r < 8; ++r)
      xout[(size_t)(row0 + r + 8 * half) * D + nt * 16 + lm] = xc[nt * 8 + r];

  // ---- single end-of-flow cross-lane reduction for logp ----
#pragma unroll
  for (int r = 0; r < 8; ++r)
    sRed[wave][r + 8 * half][lm] = lpacc[r];
  asm volatile("s_wait_dscnt 0x0" ::: "memory");
  if (lane < 16) {
    float tot = 0.f;
    const float4* rowp = (const float4*)&sRed[wave][lane][0];
#pragma unroll
    for (int q = 0; q < 4; ++q) {
      const float4 v4 = rowp[q];
      tot += v4.x + v4.y + v4.z + v4.w;
    }
    lpout[row0 + lane] = logp[row0 + lane] + (-(MAF_SIGN) * MAF_EPS) * tot;
  }
  (void)b2;  // constant offset of u: no effect on grad or laplacian
}

extern "C" void kernel_launch(void* const* d_in, const int* in_sizes, int n_in,
                              void* d_out, int out_size, void* d_ws, size_t ws_size,
                              hipStream_t stream) {
  const float* x    = (const float*)d_in[0];
  const float* logp = (const float*)d_in[1];
  const float* w1   = (const float*)d_in[2];
  const float* b1   = (const float*)d_in[3];
  const float* w2   = (const float*)d_in[4];
  const float* b2   = (const float*)d_in[5];
  float* out = (float*)d_out;

  const int B = in_sizes[1];            // 4096 samples (logp length)
  float* xout  = out;                   // [B,32] flat, then logp [B]
  float* lpout = out + (size_t)B * 32;

  dim3 grid(B / 64);                    // 16 samples/wave * 4 waves/block
  dim3 block(128);
  hipLaunchKernelGGL(maflow_rk4_wmma, grid, block, 0, stream,
                     x, logp, w1, b1, w2, b2, xout, lpout);
}
